// RelativeMultiHeadAttention_62998580298335
// MI455X (gfx1250) — matrix operand order repaired and checked by reading, not compiled, not run on hardware
//
#include <hip/hip_runtime.h>
#include <hip/hip_bf16.h>

// ---------------------------------------------------------------------------
// Transformer-XL relative multi-head attention for MI455X (gfx1250, wave32).
// All GEMM work goes through v_wmma_f32_16x16x32_f16 (f16 in, f32 accum).
// Tile staging uses CDNA5 GLOBAL_LOAD_ASYNC_TO_LDS_B128 (ASYNCcnt) instead of
// load+ds_store pairs; next-tile global_prefetch_b8 hides HBM latency.
// Flash-style online softmax; rel_shift handled via per-workgroup bd_raw
// slabs in global scratch (each WG needs only raw rows [q0, q0+128]).
//
// Workspace requirement: ~360 MB (slabs dominate: 512 WGs * 130*2048 f16).
// ---------------------------------------------------------------------------

typedef _Float16 half_t;
typedef __attribute__((ext_vector_type(16))) _Float16 v16h;
typedef __attribute__((ext_vector_type(8)))  float    v8f;

#define BB    4
#define QQ    1024
#define MEML  1024
#define DD    1024
#define HH    16
#define DHD   64
#define KL    2048   // KLEN = MEM + Q
#define SLAB_ROWS 130
#define SLAB_ELEMS (SLAB_ROWS * KL)

__device__ __forceinline__ v8f wmma_f16(v16h a, v16h b, v8f c) {
  return __builtin_amdgcn_wmma_f32_16x16x32_f16(false, a, false, b, (short)0, c,
                                                false, false);
}

// Async global->LDS 16B copy (CDNA5 TDM-adjacent path, tracked by ASYNCcnt).
// LDS destination = low 32 bits of the generic shared-pointer.
// Both addresses must be 16B aligned.
__device__ __forceinline__ void async_copy_b128(void* lds_dst, const void* gsrc) {
  uint32_t l = (uint32_t)(uintptr_t)lds_dst;
  uint64_t g = (uint64_t)(uintptr_t)gsrc;
  asm volatile("global_load_async_to_lds_b128 %0, %1, off"
               :: "v"(l), "v"(g) : "memory");
}
__device__ __forceinline__ void async_wait0() {
  asm volatile("s_wait_asynccnt 0x0" ::: "memory");
}

// A-matrix fragment (16x32 f16). base points at element (row0, k0) of a
// row-major [M][K] matrix with leading dim `ld` halves (ld % 8 == 0, 16B align).
// Lanes 0-15: row=lane, elems 0-7 = K0..7, 8-15 = K16..23.
// Lanes 16-31: row=lane-16, elems 0-7 = K8..15, 8-15 = K24..31.
__device__ __forceinline__ v16h load_frag_a(const half_t* base, int ld) {
  int lane = threadIdx.x & 31;
  const half_t* p = base + (lane & 15) * ld + ((lane & 16) ? 8 : 0);
  v16h r;
  ((uint4*)&r)[0] = *(const uint4*)(p);
  ((uint4*)&r)[1] = *(const uint4*)(p + 16);
  return r;
}

// B-matrix fragment (32x16 f16) from [N][K]-major storage; base at (n0, k0).
// Lanes 0-15: col=lane, elems 0-7 = K0..7, 8-15 = K16..23.
// Lanes 16-31: col=lane-16, elems 0-7 = K8..15, 8-15 = K24..31.
__device__ __forceinline__ v16h load_frag_b(const half_t* base, int ld) {
  int lane = threadIdx.x & 31;
  const half_t* p = base + (lane & 15) * ld + ((lane & 16) ? 8 : 0);
  v16h r;
  ((uint4*)&r)[0] = *(const uint4*)(p);
  ((uint4*)&r)[1] = *(const uint4*)(p + 16);
  return r;
}

// ----------------------------- elementwise ---------------------------------
__global__ void cvt_f16_kernel(const float* __restrict__ src,
                               half_t* __restrict__ dst, int n) {
  int i = blockIdx.x * blockDim.x + threadIdx.x;
  if (i < n) dst[i] = (half_t)src[i];
}

// 1024x1024 f32 -> f16 transpose (dst[n][k] = src[k][n])
__global__ void transpose_w_kernel(const float* __restrict__ src,
                                   half_t* __restrict__ dst) {
  __shared__ float tile[32][33];
  int bx = blockIdx.x * 32, by = blockIdx.y * 32;
  int tx = threadIdx.x, ty = threadIdx.y;  // block (32,8)
  for (int r = ty; r < 32; r += 8)
    tile[r][tx] = src[(size_t)(by + r) * DD + bx + tx];
  __syncthreads();
  for (int r = ty; r < 32; r += 8)
    dst[(size_t)(bx + r) * DD + by + tx] = (half_t)tile[tx][r];
}

// ------------------------------ projections --------------------------------
// C(MxN=Mx1024) = A(f16, Mx1024) * W^T-stored(f16, [N][K]) with epilogues.
// mode 0: head-split f16 store [Bb,H,Tdst,64] (+opt bias1/dst1, bias2/dst2)
// mode 1: transposed head-split f16 store [Bb,H,64,KL] (for V)
// mode 2: plain f32 row-major store (out projection)
// cache (optional): f32 [Bb,H,1024,64] store of raw projection (k/v caches).
__launch_bounds__(256, 1)
__global__ void gemm_proj_kernel(const half_t* __restrict__ A,
                                 const half_t* __restrict__ WT,
                                 int T, int Tdst, int t0, int mode,
                                 const float* __restrict__ bias1,
                                 half_t* __restrict__ dst1,
                                 const float* __restrict__ bias2,
                                 half_t* __restrict__ dst2,
                                 float* __restrict__ cache,
                                 float* __restrict__ outf32) {
  __shared__ half_t a_lds[128 * 40];
  __shared__ half_t w_lds[128 * 40];
  const int tid = threadIdx.x;
  const int w = tid >> 5, lane = tid & 31;
  const int m0 = blockIdx.x * 128, n0 = blockIdx.y * 128;
  const int wr = w & 3, wc = w >> 2;  // wave computes 32 rows x 64 cols
  v8f c[2][4] = {};

  for (int kk = 0; kk < DD; kk += 32) {
    __syncthreads();
    // async global -> LDS tile staging (no VGPR round-trip)
    for (int i = tid; i < 512; i += 256) {
      int row = i >> 2, ch = i & 3;
      async_copy_b128(&a_lds[row * 40 + ch * 8],
                      &A[(size_t)(m0 + row) * DD + kk + ch * 8]);
      async_copy_b128(&w_lds[row * 40 + ch * 8],
                      &WT[(size_t)(n0 + row) * DD + kk + ch * 8]);
    }
    // prefetch next k-slice while waiting
    if (kk + 32 < DD) {
      int row = tid >> 1, half = tid & 1;
      __builtin_prefetch(&A[(size_t)(m0 + row) * DD + kk + 32 + half * 16], 0, 1);
      __builtin_prefetch(&WT[(size_t)(n0 + row) * DD + kk + 32 + half * 16], 0, 1);
    }
    async_wait0();
    __syncthreads();
    v16h af0 = load_frag_a(&a_lds[(wr * 32) * 40], 40);
    v16h af1 = load_frag_a(&a_lds[(wr * 32 + 16) * 40], 40);
    v16h bf[4];
#pragma unroll
    for (int jt = 0; jt < 4; ++jt)
      bf[jt] = load_frag_b(&w_lds[(wc * 64 + jt * 16) * 40], 40);
#pragma unroll
    for (int jt = 0; jt < 4; ++jt) {
      c[0][jt] = wmma_f16(af0, bf[jt], c[0][jt]);
      c[1][jt] = wmma_f16(af1, bf[jt], c[1][jt]);
    }
  }

  const int colL = lane & 15;
  const int rhi = (lane & 16) ? 8 : 0;
#pragma unroll
  for (int i2 = 0; i2 < 2; ++i2) {
#pragma unroll
    for (int jt = 0; jt < 4; ++jt) {
#pragma unroll
      for (int j = 0; j < 8; ++j) {
        int mrow = m0 + wr * 32 + i2 * 16 + rhi + j;
        int ncol = n0 + wc * 64 + jt * 16 + colL;
        float v = c[i2][jt][j];
        if (mode == 2) {
          outf32[(size_t)mrow * DD + ncol] = v;
          continue;
        }
        int bidx = mrow / T, t = mrow % T;
        int hh = ncol >> 6, d = ncol & 63;
        if (cache)
          cache[(((size_t)(bidx * HH + hh)) * MEML + t) * DHD + d] = v;
        size_t idx;
        if (mode == 0)
          idx = (((size_t)(bidx * HH + hh)) * Tdst + t0 + t) * DHD + d;
        else  // mode 1: V transposed [b,h,dh,KL]
          idx = (((size_t)(bidx * HH + hh)) * DHD + d) * (size_t)KL + (t0 + t);
        float b1 = bias1 ? bias1[ncol] : 0.f;
        dst1[idx] = (half_t)(v + b1);
        if (dst2) dst2[idx] = (half_t)(v + bias2[ncol]);
      }
    }
  }
}

// --------------------------- fused attention -------------------------------
// Grid: 512 WGs = (b,h) x (Q/128). 256 threads = 8 waves; wave w owns 16 rows.
// Phase 1: bd_raw rows [q0, q0+129] into private global slab (WMMA).
// Phase 2: flash attention over KLEN with shifted-bd gather from the slab.
__launch_bounds__(256, 1)
__global__ void attn_kernel(const half_t* __restrict__ qw,
                            const half_t* __restrict__ qr,
                            const half_t* __restrict__ kall,
                            const half_t* __restrict__ vT,
                            const half_t* __restrict__ rhead,
                            half_t* __restrict__ attn,
                            half_t* __restrict__ slabs) {
  __shared__ half_t k_lds[128 * 72];      // [token][dh], ld 72
  __shared__ half_t v_lds[64 * 136];      // [dh][token], ld 136
  __shared__ half_t p_lds[8 * 16 * 136];  // per-wave 16x128 prob tile

  const int wg = blockIdx.x;
  const int qblk = wg & 7, bh = wg >> 3;
  const int b = bh >> 4, h = bh & 15;
  const int q0 = qblk * 128;
  const int tid = threadIdx.x;
  const int w = tid >> 5, lane = tid & 31;
  const int colL = lane & 15;
  const int rhi = (lane & 16) ? 8 : 0;

  half_t* slab = slabs + (size_t)wg * SLAB_ELEMS;

  // ---------------- Phase 1: bd_raw = (q + r_r_bias) @ r_head^T -----------
  const half_t* qr_base = qr + ((size_t)bh * QQ + q0) * DHD;
  const half_t* rh_base = rhead + (size_t)h * KL * DHD;
  for (int rt = w; rt < 9; rt += 8) {  // 9 row-tiles cover 129 needed rows
    v16h a0 = load_frag_a(qr_base + rt * 16 * DHD, DHD);
    v16h a1 = load_frag_a(qr_base + rt * 16 * DHD + 32, DHD);
    for (int n0 = 0; n0 < KL; n0 += 16) {
      v16h b0 = load_frag_b(rh_base + (size_t)n0 * DHD, DHD);
      v16h b1 = load_frag_b(rh_base + (size_t)n0 * DHD + 32, DHD);
      v8f c = {};
      c = wmma_f16(a0, b0, c);
      c = wmma_f16(a1, b1, c);
      int col = n0 + colL;
#pragma unroll
      for (int j = 0; j < 8; ++j) {
        int row = rt * 16 + rhi + j;
        if (row < SLAB_ROWS) slab[(size_t)row * KL + col] = (half_t)c[j];
      }
    }
  }
  __threadfence();  // slab visible to all waves in this WG
  __syncthreads();

  // ---------------- Phase 2: flash attention ------------------------------
  const half_t* qw_base = qw + ((size_t)bh * QQ + q0 + w * 16) * DHD;
  v16h qa0 = load_frag_a(qw_base, DHD);
  v16h qa1 = load_frag_a(qw_base + 32, DHD);
  const half_t* k_base = kall + (size_t)bh * KL * DHD;
  const half_t* v_base = vT + (size_t)bh * DHD * KL;
  half_t* pw = &p_lds[w * 16 * 136];

  float mrow[8], lrow[8];
  v8f oacc[4] = {};
#pragma unroll
  for (int j = 0; j < 8; ++j) { mrow[j] = -1e30f; lrow[j] = 0.f; }

  for (int kt = 0; kt < KL; kt += 128) {
    __syncthreads();
    // async global -> LDS staging of K and V^T tiles
    for (int i = tid; i < 1024; i += 256) {  // K tile [128][64] -> ld 72
      int row = i >> 3, ch = i & 7;
      async_copy_b128(&k_lds[row * 72 + ch * 8],
                      &k_base[(size_t)(kt + row) * DHD + ch * 8]);
    }
    for (int i = tid; i < 1024; i += 256) {  // V^T tile [64][128] -> ld 136
      int row = i >> 4, ch = i & 15;
      async_copy_b128(&v_lds[row * 136 + ch * 8],
                      &v_base[(size_t)row * KL + kt + ch * 8]);
    }
    // prefetch next tiles into L2 while async copies land
    if (kt + 128 < KL) {
      __builtin_prefetch(&k_base[(size_t)(kt + 128) * DHD + tid * 32], 0, 1);
      if (tid < 64)
        __builtin_prefetch(&v_base[(size_t)tid * KL + kt + 128], 0, 1);
    }
    async_wait0();
    __syncthreads();

    // ac = (q + r_w_bias) @ K^T for this 16x128 tile
    v8f s[8];
#pragma unroll
    for (int nt = 0; nt < 8; ++nt) {
      v16h b0 = load_frag_b(&k_lds[(nt * 16) * 72], 72);
      v16h b1 = load_frag_b(&k_lds[(nt * 16) * 72 + 32], 72);
      v8f c = {};
      c = wmma_f16(qa0, b0, c);
      c = wmma_f16(qa1, b1, c);
      s[nt] = c;
    }

    // add rel-shifted bd, scale by 1/sqrt(DH)
#pragma unroll
    for (int nt = 0; nt < 8; ++nt) {
#pragma unroll
      for (int j = 0; j < 8; ++j) {
        int rl = w * 16 + rhi + j;       // local q row in [0,128)
        int kj = kt + nt * 16 + colL;
        int qi = q0 + rl;
        int mm = kj + (QQ - 1) - qi;     // raw column index
        float bdv;
        if (mm < KL)
          bdv = (float)slab[(size_t)rl * KL + mm];
        else if (mm == KL)
          bdv = 0.f;                     // the pad zero
        else
          bdv = (float)slab[(size_t)(rl + 1) * KL + (mm - (KL + 1))];
        s[nt][j] = (s[nt][j] + bdv) * 0.125f;
      }
    }

    // online softmax (row stats live replicated per 16-lane half)
    float mnew[8], sc[8], rsum[8];
#pragma unroll
    for (int j = 0; j < 8; ++j) {
      float mx = s[0][j];
#pragma unroll
      for (int nt = 1; nt < 8; ++nt) mx = fmaxf(mx, s[nt][j]);
#pragma unroll
      for (int off = 1; off < 16; off <<= 1)
        mx = fmaxf(mx, __shfl_xor(mx, off, 32));
      mnew[j] = fmaxf(mrow[j], mx);
      rsum[j] = 0.f;
    }
#pragma unroll
    for (int nt = 0; nt < 8; ++nt) {
#pragma unroll
      for (int j = 0; j < 8; ++j) {
        float p = __expf(s[nt][j] - mnew[j]);
        s[nt][j] = p;
        rsum[j] += p;
      }
    }
#pragma unroll
    for (int j = 0; j < 8; ++j) {
#pragma unroll
      for (int off = 1; off < 16; off <<= 1)
        rsum[j] += __shfl_xor(rsum[j], off, 32);
      sc[j] = __expf(mrow[j] - mnew[j]);
      lrow[j] = lrow[j] * sc[j] + rsum[j];
      mrow[j] = mnew[j];
    }
#pragma unroll
    for (int ot = 0; ot < 4; ++ot)
#pragma unroll
      for (int j = 0; j < 8; ++j) oacc[ot][j] *= sc[j];

    // stage probs to wave-private LDS and do P @ V via WMMA
#pragma unroll
    for (int nt = 0; nt < 8; ++nt)
#pragma unroll
      for (int j = 0; j < 8; ++j)
        pw[(rhi + j) * 136 + nt * 16 + colL] = (half_t)s[nt][j];

#pragma unroll
    for (int ot = 0; ot < 4; ++ot) {
#pragma unroll
      for (int ks = 0; ks < 4; ++ks) {
        v16h a = load_frag_a(&pw[ks * 32], 136);
        v16h bfr = load_frag_b(&v_lds[(ot * 16) * 136 + ks * 32], 136);
        oacc[ot] = wmma_f16(a, bfr, oacc[ot]);
      }
    }
  }

  // write attn_vec as f16 [B, Q, H*DH] (A matrix of out projection)
#pragma unroll
  for (int ot = 0; ot < 4; ++ot) {
#pragma unroll
    for (int j = 0; j < 8; ++j) {
      int qi = q0 + w * 16 + rhi + j;
      int col = h * DHD + ot * 16 + colL;
      attn[((size_t)(b * QQ + qi)) * DD + col] = (half_t)(oacc[ot][j] / lrow[j]);
    }
  }
}

// ------------------------------- launcher ----------------------------------
extern "C" void kernel_launch(void* const* d_in, const int* in_sizes, int n_in,
                              void* d_out, int out_size, void* d_ws,
                              size_t ws_size, hipStream_t stream) {
  const float* f_h   = (const float*)d_in[0];
  const float* f_mem = (const float*)d_in[1];
  const float* f_r   = (const float*)d_in[2];
  const float* f_Wq  = (const float*)d_in[3];
  const float* f_Wk  = (const float*)d_in[4];
  const float* f_Wv  = (const float*)d_in[5];
  const float* f_Wr  = (const float*)d_in[6];
  const float* f_Wo  = (const float*)d_in[7];
  const float* f_rwb = (const float*)d_in[8];
  const float* f_rrb = (const float*)d_in[9];

  float* out_f32 = (float*)d_out;                       // [B,Q,D]
  float* kcache  = out_f32 + (size_t)BB * QQ * DD;      // [B,H,MEM,DH]
  float* vcache  = kcache + (size_t)BB * HH * MEML * DHD;

  // workspace carve-up (half units)
  half_t* p = (half_t*)d_ws;
  const size_t PAD = 144 * DHD;  // phase-1 row-tile overrun padding
  half_t* h16   = p; p += (size_t)BB * QQ * DD;
  half_t* mem16 = p; p += (size_t)BB * MEML * DD;
  half_t* r16   = p; p += (size_t)KL * DD;
  half_t* WqT   = p; p += (size_t)DD * DD;
  half_t* WkT   = p; p += (size_t)DD * DD;
  half_t* WvT   = p; p += (size_t)DD * DD;
  half_t* WrT   = p; p += (size_t)DD * DD;
  half_t* WoT   = p; p += (size_t)DD * DD;
  half_t* qw16  = p; p += (size_t)BB * HH * QQ * DHD + PAD;
  half_t* qr16  = p; p += (size_t)BB * HH * QQ * DHD + PAD;
  half_t* kall  = p; p += (size_t)BB * HH * KL * DHD;
  half_t* vT16  = p; p += (size_t)BB * HH * DHD * KL;
  half_t* rhd   = p; p += (size_t)HH * KL * DHD;
  half_t* attn  = p; p += (size_t)BB * QQ * DD;
  half_t* slabs = p;  // 512 * SLAB_ELEMS halves (~272 MB)

  // 1) f32 -> f16 activations
  cvt_f16_kernel<<<(BB * QQ * DD + 255) / 256, 256, 0, stream>>>(f_h, h16, BB * QQ * DD);
  cvt_f16_kernel<<<(BB * MEML * DD + 255) / 256, 256, 0, stream>>>(f_mem, mem16, BB * MEML * DD);
  cvt_f16_kernel<<<(KL * DD + 255) / 256, 256, 0, stream>>>(f_r, r16, KL * DD);

  // 2) weight transposes to [N][K] f16
  dim3 tb(32, 8), tg(DD / 32, DD / 32);
  transpose_w_kernel<<<tg, tb, 0, stream>>>(f_Wq, WqT);
  transpose_w_kernel<<<tg, tb, 0, stream>>>(f_Wk, WkT);
  transpose_w_kernel<<<tg, tb, 0, stream>>>(f_Wv, WvT);
  transpose_w_kernel<<<tg, tb, 0, stream>>>(f_Wr, WrT);
  transpose_w_kernel<<<tg, tb, 0, stream>>>(f_Wo, WoT);

  // 3) projection GEMMs (WMMA)
  dim3 g4k(BB * QQ / 128, DD / 128);   // M=4096
  dim3 g2k(KL / 128, DD / 128);        // M=2048
  // q -> qw (= q + r_w_bias) and qr (= q + r_r_bias), head-split
  gemm_proj_kernel<<<g4k, 256, 0, stream>>>(h16, WqT, QQ, QQ, 0, 0,
                                            f_rwb, qw16, f_rrb, qr16, nullptr, nullptr);
  // k_mem -> k_all[:, :, 0:1024]
  gemm_proj_kernel<<<g4k, 256, 0, stream>>>(mem16, WkT, MEML, KL, 0, 0,
                                            nullptr, kall, nullptr, nullptr, nullptr, nullptr);
  // k_t -> k_all[:, :, 1024:2048] + f32 k-cache
  gemm_proj_kernel<<<g4k, 256, 0, stream>>>(h16, WkT, QQ, KL, MEML, 0,
                                            nullptr, kall, nullptr, nullptr, kcache, nullptr);
  // v_mem -> vT[:, :, :, 0:1024]
  gemm_proj_kernel<<<g4k, 256, 0, stream>>>(mem16, WvT, MEML, KL, 0, 1,
                                            nullptr, vT16, nullptr, nullptr, nullptr, nullptr);
  // v_t -> vT[:, :, :, 1024:2048] + f32 v-cache
  gemm_proj_kernel<<<g4k, 256, 0, stream>>>(h16, WvT, QQ, KL, MEML, 1,
                                            nullptr, vT16, nullptr, nullptr, vcache, nullptr);
  // r -> r_head [H, KLEN, DH]
  gemm_proj_kernel<<<g2k, 256, 0, stream>>>(r16, WrT, KL, KL, 0, 0,
                                            nullptr, rhd, nullptr, nullptr, nullptr, nullptr);

  // 4) fused rel-shift + flash attention
  attn_kernel<<<BB * HH * (QQ / 128), 256, 0, stream>>>(qw16, qr16, kall, vT16,
                                                        rhd, attn, slabs);

  // 5) out = attn_vec @ Wo (f32 into d_out)
  gemm_proj_kernel<<<g4k, 256, 0, stream>>>(attn, WoT, QQ, 0, 0, 2,
                                            nullptr, nullptr, nullptr, nullptr,
                                            nullptr, out_f32);
}
